// UseAllConvNet_37134287241670
// MI455X (gfx1250) — matrix operand order repaired and checked by reading, not compiled, not run on hardware
//
#include <hip/hip_runtime.h>
#include <math.h>

typedef __attribute__((ext_vector_type(16))) _Float16 v16h;
typedef __attribute__((ext_vector_type(8)))  _Float16 v8h;
typedef __attribute__((ext_vector_type(8)))  float    v8f;

#define BATCH 8
#define NPTS  2048
#define KK    16
#define VG    40
#define VSZ   0.05f
#define PIF   3.14159265358979323846f

// ---------------- CDNA5 async global->LDS helpers ----------------
__device__ __forceinline__ unsigned lds_off_u32(void* p) {
  // generic -> LDS(addrspace 3) cast; as3 pointer value IS the wave LDS byte offset
  return (unsigned)(unsigned long long)(__attribute__((address_space(3))) char*)p;
}

__device__ __forceinline__ void async_g2l_b128(unsigned ldsoff, const void* gsrc) {
  asm volatile("global_load_async_to_lds_b128 %0, %1, off"
               :: "v"(ldsoff), "v"(gsrc)
               : "memory");
}

__device__ __forceinline__ void wait_async0() {
#if __has_builtin(__builtin_amdgcn_s_wait_asynccnt)
  __builtin_amdgcn_s_wait_asynccnt(0);
#else
  asm volatile("s_wait_asynccnt 0x0" ::: "memory");
#endif
}

// ---------------- shared-memory reduction ----------------
__device__ __forceinline__ float blk_sum(float v, float* buf, int t, int nthr) {
  buf[t] = v;
  __syncthreads();
  for (int s = nthr >> 1; s > 0; s >>= 1) {
    if (t < s) buf[t] += buf[t + s];
    __syncthreads();
  }
  float r = buf[0];
  __syncthreads();
  return r;
}

// ---------------- f32 [K,Co] -> f16 transposed [Co,K] ----------------
__global__ void k_cvt_f16_t(const float* __restrict__ s, _Float16* __restrict__ d,
                            int K, int Co) {
  int i = blockIdx.x * blockDim.x + threadIdx.x;
  if (i >= K * Co) return;
  int k = i / Co, n = i % Co;
  d[(size_t)n * K + k] = (_Float16)s[i];
}

// ---------------- positional encoding: [B,N,3] -> [B,N,24] ----------------
__global__ void k_posenc(const float* __restrict__ x, float* __restrict__ f0) {
  int p = blockIdx.x * blockDim.x + threadIdx.x;
  if (p >= BATCH * NPTS) return;
  const float* xp = x + p * 3;
  float* op = f0 + p * 24;
#pragma unroll
  for (int c = 0; c < 3; ++c) {
    float v = xp[c];
    float fr = PIF;
#pragma unroll
    for (int j = 0; j < 4; ++j) {
      float a = v * fr;
      op[c * 8 + j]     = sinf(a);
      op[c * 8 + 4 + j] = cosf(a);
      fr *= 2.0f;
    }
  }
}

// ---------------- brute-force knn(16), per batch ----------------
__global__ void k_knn(const float* __restrict__ x, int* __restrict__ kidx,
                      float* __restrict__ kd2) {
  int p = blockIdx.x * blockDim.x + threadIdx.x;
  if (p >= BATCH * NPTS) return;
  int b = p / NPTS;
  const float* xb = x + (size_t)b * NPTS * 3;
  float qx = x[p * 3 + 0], qy = x[p * 3 + 1], qz = x[p * 3 + 2];
  float bd[KK];
  int   bi[KK];
#pragma unroll
  for (int t = 0; t < KK; ++t) { bd[t] = 3.0e38f; bi[t] = 0; }
  int worst = 0;
  float wd = 3.0e38f;
  for (int j = 0; j < NPTS; ++j) {
    float dx = qx - xb[j * 3 + 0];
    float dy = qy - xb[j * 3 + 1];
    float dz = qz - xb[j * 3 + 2];
    float d = dx * dx + dy * dy + dz * dz;
    if (d < wd) {
      bd[worst] = d; bi[worst] = j;
      wd = bd[0]; worst = 0;
#pragma unroll
      for (int t = 1; t < KK; ++t)
        if (bd[t] > wd) { wd = bd[t]; worst = t; }
    }
  }
#pragma unroll
  for (int t = 0; t < KK; ++t) {
    kidx[(size_t)p * KK + t] = bi[t];
    kd2[(size_t)p * KK + t]  = fmaxf(bd[t], 0.0f);
  }
}

// ---------------- PointConv1: knn mean agg -> 24x64 linear, LN, ReLU ----------------
__global__ void k_pc1(const float* __restrict__ f0, const int* __restrict__ kidx,
                      const float* __restrict__ W, const float* __restrict__ bias,
                      const float* __restrict__ g, const float* __restrict__ be,
                      float* __restrict__ f1) {
  __shared__ float agg[24];
  __shared__ float red[64];
  int p = blockIdx.x, t = threadIdx.x;
  int b = p / NPTS;
  if (t < 24) {
    float s = 0.f;
    for (int k = 0; k < KK; ++k) {
      int nb = kidx[(size_t)p * KK + k];
      s += f0[((size_t)b * NPTS + nb) * 24 + t];
    }
    agg[t] = s * (1.0f / KK);
  }
  __syncthreads();
  float y = bias[t];
#pragma unroll
  for (int i = 0; i < 24; ++i) y += agg[i] * W[i * 64 + t];
  float mean = blk_sum(y, red, t, 64) * (1.0f / 64.0f);
  float d = y - mean;
  float var = blk_sum(d * d, red, t, 64) * (1.0f / 64.0f);
  float o = d * rsqrtf(var + 1e-5f) * g[t] + be[t];
  f1[(size_t)p * 64 + t] = fmaxf(o, 0.f);
}

// ---------------- PointConv2: radius-masked mean -> 64x64 linear, LN, ReLU ----------------
__global__ void k_pc2(const float* __restrict__ f1, const int* __restrict__ kidx,
                      const float* __restrict__ kd2,
                      const float* __restrict__ W, const float* __restrict__ bias,
                      const float* __restrict__ g, const float* __restrict__ be,
                      float* __restrict__ f2) {
  __shared__ float agg[64];
  __shared__ float red[64];
  int p = blockIdx.x, t = threadIdx.x;
  int b = p / NPTS;
  float s = 0.f, cn = 0.f;
  for (int k = 0; k < KK; ++k) {
    float m = (kd2[(size_t)p * KK + k] <= VSZ * VSZ) ? 1.f : 0.f;
    cn += m;
    int nb = kidx[(size_t)p * KK + k];
    s += m * f1[((size_t)b * NPTS + nb) * 64 + t];
  }
  agg[t] = s / fmaxf(cn, 1.f);
  __syncthreads();
  float y = bias[t];
#pragma unroll
  for (int i = 0; i < 64; ++i) y += agg[i] * W[i * 64 + t];
  float mean = blk_sum(y, red, t, 64) * (1.0f / 64.0f);
  float d = y - mean;
  float var = blk_sum(d * d, red, t, 64) * (1.0f / 64.0f);
  float o = d * rsqrtf(var + 1e-5f) * g[t] + be[t];
  f2[(size_t)p * 64 + t] = fmaxf(o, 0.f);
}

// ---------------- voxelize: scatter-add into 40^3 grid ----------------
__global__ void k_voxelize(const float* __restrict__ x, const float* __restrict__ f2,
                           float* __restrict__ vsum, float* __restrict__ cnt) {
  int p = blockIdx.x, t = threadIdx.x;
  int b = p / NPTS;
  int v0 = min(max((int)floorf(x[p * 3 + 0] / VSZ), -VG / 2), VG / 2 - 1) + VG / 2;
  int v1 = min(max((int)floorf(x[p * 3 + 1] / VSZ), -VG / 2), VG / 2 - 1) + VG / 2;
  int v2 = min(max((int)floorf(x[p * 3 + 2] / VSZ), -VG / 2), VG / 2 - 1) + VG / 2;
  int flat = ((b * VG + v0) * VG + v1) * VG + v2;
  atomicAdd(&vsum[(size_t)flat * 64 + t], f2[(size_t)p * 64 + t]);
  if (t == 0) atomicAdd(&cnt[flat], 1.f);
}

__global__ void k_vox_final(float* __restrict__ v, const float* __restrict__ cnt,
                            float* __restrict__ mask, int nvox) {
  int i = blockIdx.x * blockDim.x + threadIdx.x;
  if (i >= nvox * 64) return;
  int vx = i >> 6;
  float c = cnt[vx];
  v[i] = v[i] / fmaxf(c, 1.f);
  if ((i & 63) == 0) mask[vx] = (c > 0.f) ? 1.f : 0.f;
}

// ---------------- implicit-GEMM 3D conv, double-buffered async-LDS WMMA pipeline -------
// Workgroup: 256 threads = 8 waves, computes a 32(M-voxels) x 64(Co) tile.
// Weights pre-transposed to Bt[Co][K] => the 64x32 weight tile is layout-identical in
// global and LDS, staged with GLOBAL_LOAD_ASYNC_TO_LDS_B128 (ASYNCcnt).
// Double-buffered: chunk i+1's async B copy + A global load are issued right after the
// single per-iteration barrier, overlapping with chunk i's WMMA.
__global__ void __launch_bounds__(256)
k_conv_wmma(const float* __restrict__ in, const _Float16* __restrict__ bt16,
            const float* __restrict__ bias, float* __restrict__ out,
            int Din, int Dout, int Ci, int Co, int ks, int stride, int pad) {
  __shared__ _Float16 aSm[2][32][32];
  __shared__ _Float16 bT[2][64][32];
  const int tid  = threadIdx.x;
  const int lane = tid & 31;
  const int wv   = tid >> 5;       // 0..7
  const int mw   = wv & 1;         // M subtile 0..1
  const int nw   = wv >> 1;        // N subtile 0..3
  const int dvol = Dout * Dout * Dout;
  const int M = BATCH * dvol;
  const int mbase = blockIdx.x * 32;
  const int nbase = blockIdx.y * 64;
  // --- staging roles ---
  const int ar = tid >> 3;         // A: row 0..31
  const int aq = (tid & 7) * 4;    // A: 4-float quad within row
  const int mA = mbase + ar;
  const int b  = mA / dvol;
  const int rv = mA % dvol;
  const int oz = rv / (Dout * Dout);
  const int oy = (rv / Dout) % Dout;
  const int ox = rv % Dout;
  const bool rowok = (mA < M);
  const int bnr = tid >> 2;        // B: n-row 0..63
  const int bc  = (tid & 3) * 8;   // B: k offset 0,8,16,24 (16B chunk)
  const unsigned bLds0 = lds_off_u32(&bT[0][bnr][bc]);
  const unsigned bLds1 = lds_off_u32(&bT[1][bnr][bc]);
  // --- fragment roles (ISA 7.12.2 wave32 layouts) ---
  const int lo  = (lane < 16) ? 1 : 0;
  const int kA0 = lo ? 0 : 8, kA1 = lo ? 16 : 24, kB0 = lo ? 0 : 16;
  const int frow = (lane & 15) + mw * 16;
  const int fn   = (lane & 15) + nw * 16;
  const int K = ks * ks * ks * Ci;
  const int nch = K / 32;
  const _Float16* bRow = bt16 + (size_t)(nbase + bnr) * K + bc;

  // per-chunk A gather (f32x4, zero-filled outside the padded volume)
  auto load_a = [&](int kb) -> float4 {
    const int tap = kb / Ci;
    const int ci0 = kb % Ci;
    const int kz = tap / (ks * ks);
    const int ky = (tap / ks) % ks;
    const int kx = tap % ks;
    const int iz = oz * stride + kz - pad;
    const int iy = oy * stride + ky - pad;
    const int ix = ox * stride + kx - pad;
    if (rowok && iz >= 0 && iz < Din && iy >= 0 && iy < Din && ix >= 0 && ix < Din)
      return *(const float4*)(
          in + ((((size_t)b * Din + iz) * Din + iy) * Din + ix) * Ci + ci0 + aq);
    return make_float4(0.f, 0.f, 0.f, 0.f);
  };

  v8f acc = {};
  // prologue: chunk 0 in flight
  async_g2l_b128(bLds0, bRow);
  float4 regA = load_a(0);
  for (int i = 0; i < nch; ++i) {
    const int buf = i & 1;
    // commit pre-loaded A registers to LDS (f32 -> f16)
    aSm[buf][ar][aq + 0] = (_Float16)regA.x;
    aSm[buf][ar][aq + 1] = (_Float16)regA.y;
    aSm[buf][ar][aq + 2] = (_Float16)regA.z;
    aSm[buf][ar][aq + 3] = (_Float16)regA.w;
    wait_async0();               // chunk i's B tile landed in bT[buf]
    __syncthreads();             // one barrier per chunk (double buffering)
    if (i + 1 < nch) {           // overlap chunk i+1's movement with chunk i's WMMA
      async_g2l_b128(buf ? bLds0 : bLds1, bRow + (size_t)(i + 1) * 32);
      regA = load_a((i + 1) * 32);
    }
    // fragments from LDS: two b128 loads each
    v8h a0p = *(const v8h*)&aSm[buf][frow][kA0];
    v8h a1p = *(const v8h*)&aSm[buf][frow][kA1];
    v8h b0p = *(const v8h*)&bT[buf][fn][kA0];
    v8h b1p = *(const v8h*)&bT[buf][fn][kA1];
    v16h av = __builtin_shufflevector(a0p, a1p, 0, 1, 2, 3, 4, 5, 6, 7,
                                      8, 9, 10, 11, 12, 13, 14, 15);
    v16h bw = __builtin_shufflevector(b0p, b1p, 0, 1, 2, 3, 4, 5, 6, 7,
                                      8, 9, 10, 11, 12, 13, 14, 15);
    acc = __builtin_amdgcn_wmma_f32_16x16x32_f16(false, av, false, bw, (short)0, acc,
                                                 false, false);
  }
#pragma unroll
  for (int rr = 0; rr < 8; ++rr) {
    int mm = mbase + mw * 16 + rr + (lo ? 0 : 8);
    if (mm < M) out[(size_t)mm * Co + nbase + fn] = acc[rr] + bias[nbase + fn];
  }
}

// ---------------- activity-mask dilation (== reference ones-conv > 0) ----------------
__global__ void k_mask_dilate(const float* __restrict__ mi, float* __restrict__ mo,
                              int Din, int Dout, int ks, int stride, int pad) {
  int i = blockIdx.x * blockDim.x + threadIdx.x;
  int dvol = Dout * Dout * Dout;
  if (i >= BATCH * dvol) return;
  int b = i / dvol;
  int r = i % dvol;
  int oz = r / (Dout * Dout), oy = (r / Dout) % Dout, ox = r % Dout;
  float s = 0.f;
  for (int kz = 0; kz < ks; ++kz)
    for (int ky = 0; ky < ks; ++ky)
      for (int kx = 0; kx < ks; ++kx) {
        int iz = oz * stride + kz - pad;
        int iy = oy * stride + ky - pad;
        int ix = ox * stride + kx - pad;
        if (iz >= 0 && iz < Din && iy >= 0 && iy < Din && ix >= 0 && ix < Din)
          s += mi[(((size_t)b * Din + iz) * Din + iy) * Din + ix];
      }
  mo[i] = (s > 0.f) ? 1.f : 0.f;
}

// ---------------- per-voxel LayerNorm + ReLU, masked (in place) ----------------
__global__ void k_ln_relu_mask(float* __restrict__ y, const float* __restrict__ mask,
                               const float* __restrict__ g, const float* __restrict__ be,
                               int Co) {
  __shared__ float red[128];
  int vox = blockIdx.x, t = threadIdx.x;
  float mo = mask[vox];            // uniform per block
  float* yp = y + (size_t)vox * Co;
  if (mo == 0.f) {
    for (int c = t; c < Co; c += 128) yp[c] = 0.f;
    return;
  }
  float s = 0.f;
  for (int c = t; c < Co; c += 128) s += yp[c];
  float mean = blk_sum(s, red, t, 128) / (float)Co;
  float vs = 0.f;
  for (int c = t; c < Co; c += 128) { float d = yp[c] - mean; vs += d * d; }
  float var = blk_sum(vs, red, t, 128) / (float)Co;
  float rstd = rsqrtf(var + 1e-5f);
  for (int c = t; c < Co; c += 128)
    yp[c] = fmaxf((yp[c] - mean) * rstd * g[c] + be[c], 0.f);
}

// ---------------- training-mode BatchNorm: stats + apply(+ReLU) ----------------
__global__ void k_bn_stats(const float* __restrict__ xb, float* __restrict__ mv,
                           int Nsamp, int C) {
  __shared__ float r1[256];
  __shared__ float r2[256];
  int c = blockIdx.x, t = threadIdx.x;
  float s = 0.f, s2 = 0.f;
  for (int i = t; i < Nsamp; i += 256) {
    float v = xb[(size_t)i * C + c];
    s += v; s2 += v * v;
  }
  s  = blk_sum(s, r1, t, 256);
  s2 = blk_sum(s2, r2, t, 256);
  if (t == 0) {
    float m = s / (float)Nsamp;
    mv[c] = m;
    mv[C + c] = fmaxf(s2 / (float)Nsamp - m * m, 0.f);
  }
}

__global__ void k_bn_apply_relu(float* __restrict__ xb, const float* __restrict__ mv,
                                const float* __restrict__ g, const float* __restrict__ be,
                                int Nsamp, int C) {
  int i = blockIdx.x * blockDim.x + threadIdx.x;
  if (i >= Nsamp * C) return;
  int c = i % C;
  float o = (xb[i] - mv[c]) * rsqrtf(mv[C + c] + 1e-5f) * g[c] + be[c];
  xb[i] = fmaxf(o, 0.f);
}

// ---------------- WMMA GEMM for fc1 (M<=16, zero-padded rows; Bt[N][K]) ----------------
__global__ void k_gemm_wmma(const float* __restrict__ A, const _Float16* __restrict__ Bt,
                            const float* __restrict__ bias, float* __restrict__ out,
                            int M, int K, int N) {
  const int lane = threadIdx.x;
  const int lo = (lane < 16) ? 1 : 0;
  const int n = blockIdx.x * 16 + (lane & 15);
  const int row = lane & 15;
  const int kA0 = lo ? 0 : 8, kA1 = lo ? 16 : 24, kB0 = lo ? 0 : 16;
  const bool rok = row < M;
  const _Float16* bRow = Bt + (size_t)n * K;
  v8f acc = {};
  for (int kb = 0; kb < K; kb += 32) {
    v16h a = {};
    if (rok) {
      const float* ap = A + (size_t)row * K + kb;
      const float4* p0 = (const float4*)(ap + kA0);
      const float4* p1 = (const float4*)(ap + kA1);
      float4 q0 = p0[0], q1 = p0[1], q2 = p1[0], q3 = p1[1];
      a[0]  = (_Float16)q0.x; a[1]  = (_Float16)q0.y; a[2]  = (_Float16)q0.z; a[3]  = (_Float16)q0.w;
      a[4]  = (_Float16)q1.x; a[5]  = (_Float16)q1.y; a[6]  = (_Float16)q1.z; a[7]  = (_Float16)q1.w;
      a[8]  = (_Float16)q2.x; a[9]  = (_Float16)q2.y; a[10] = (_Float16)q2.z; a[11] = (_Float16)q2.w;
      a[12] = (_Float16)q3.x; a[13] = (_Float16)q3.y; a[14] = (_Float16)q3.z; a[15] = (_Float16)q3.w;
    }
    v8h b0p = *(const v8h*)(bRow + kb + kA0);
    v8h b1p = *(const v8h*)(bRow + kb + kA1);
    v16h bw = __builtin_shufflevector(b0p, b1p, 0, 1, 2, 3, 4, 5, 6, 7,
                                      8, 9, 10, 11, 12, 13, 14, 15);
    acc = __builtin_amdgcn_wmma_f32_16x16x32_f16(false, a, false, bw, (short)0, acc,
                                                 false, false);
  }
#pragma unroll
  for (int rr = 0; rr < 8; ++rr) {
    int mm = rr + (lo ? 0 : 8);
    if (mm < M) out[(size_t)mm * N + n] = acc[rr] + bias[n];
  }
}

// ---------------- fc2: [8,1024] @ [1024,40] ----------------
__global__ void k_fc2(const float* __restrict__ xin, const float* __restrict__ W,
                      const float* __restrict__ bias, float* __restrict__ out) {
  int i = blockIdx.x * blockDim.x + threadIdx.x;
  if (i >= BATCH * 40) return;
  int b = i / 40, c = i % 40;
  float s = bias[c];
  for (int k = 0; k < 1024; ++k) s += xin[(size_t)b * 1024 + k] * W[k * 40 + c];
  out[i] = s;
}

extern "C" void kernel_launch(void* const* d_in, const int* in_sizes, int n_in,
                              void* d_out, int out_size, void* d_ws, size_t ws_size,
                              hipStream_t stream) {
  (void)in_sizes; (void)n_in; (void)out_size; (void)ws_size;
  const float* X    = (const float*)d_in[0];
  const float* PC1W = (const float*)d_in[1];
  const float* PC1B = (const float*)d_in[2];
  const float* LN1G = (const float*)d_in[3];
  const float* LN1B = (const float*)d_in[4];
  const float* PC2W = (const float*)d_in[5];
  const float* PC2B = (const float*)d_in[6];
  const float* LN2G = (const float*)d_in[7];
  const float* LN2B = (const float*)d_in[8];
  const float* SCW[5]; const float* SCB[5]; const float* SLG[5]; const float* SLB[5];
  for (int i = 0; i < 5; ++i) {
    SCW[i] = (const float*)d_in[9 + 4 * i];
    SCB[i] = (const float*)d_in[10 + 4 * i];
    SLG[i] = (const float*)d_in[11 + 4 * i];
    SLB[i] = (const float*)d_in[12 + 4 * i];
  }
  const float* DC1W = (const float*)d_in[29]; const float* DC1B = (const float*)d_in[30];
  const float* BN1G = (const float*)d_in[31]; const float* BN1B = (const float*)d_in[32];
  const float* DC2W = (const float*)d_in[33]; const float* DC2B = (const float*)d_in[34];
  const float* BN2G = (const float*)d_in[35]; const float* BN2B = (const float*)d_in[36];
  const float* FC1W = (const float*)d_in[37]; const float* FC1B = (const float*)d_in[38];
  const float* BN3G = (const float*)d_in[39]; const float* BN3B = (const float*)d_in[40];
  const float* FC2W = (const float*)d_in[41]; const float* FC2B = (const float*)d_in[42];

  char* base = (char*)d_ws;
  size_t off = 0;
  auto alloc = [&](size_t bytes) -> void* {
    void* p = (void*)(base + off);
    off += (bytes + 255) & ~(size_t)255;
    return p;
  };

  const int NP = BATCH * NPTS;
  float* f0   = (float*)alloc((size_t)NP * 24 * 4);
  int*   kidx = (int*)  alloc((size_t)NP * KK * 4);
  float* kd2  = (float*)alloc((size_t)NP * KK * 4);
  float* f1   = (float*)alloc((size_t)NP * 64 * 4);
  float* f2   = (float*)alloc((size_t)NP * 64 * 4);
  const int NV40 = BATCH * VG * VG * VG;   // 512000
  float* v40  = (float*)alloc((size_t)NV40 * 64 * 4);
  float* y40  = (float*)alloc((size_t)NV40 * 64 * 4);
  float* cnt  = (float*)alloc((size_t)NV40 * 4);
  float* m40  = (float*)alloc((size_t)NV40 * 4);
  float* m40b = (float*)alloc((size_t)NV40 * 4);
  const int NV20 = BATCH * 20 * 20 * 20;   // 64000
  float* y20  = (float*)alloc((size_t)NV20 * 64 * 4);
  float* y20b = (float*)alloc((size_t)NV20 * 128 * 4);
  float* m20a = (float*)alloc((size_t)NV20 * 4);
  float* m20b = (float*)alloc((size_t)NV20 * 4);
  const int NV10 = BATCH * 10 * 10 * 10;   // 8000
  float* y10a = (float*)alloc((size_t)NV10 * 256 * 4);
  float* y10b = (float*)alloc((size_t)NV10 * 512 * 4);
  float* m10a = (float*)alloc((size_t)NV10 * 4);
  float* m10b = (float*)alloc((size_t)NV10 * 4);
  const int ND1 = BATCH * 125;             // 1000 samples, 1024 ch
  float* yd1  = (float*)alloc((size_t)ND1 * 1024 * 4);
  const int ND2 = BATCH * 8;               // 64 samples, 1024 ch
  float* yd2  = (float*)alloc((size_t)ND2 * 1024 * 4);
  float* yf1  = (float*)alloc((size_t)BATCH * 1024 * 4);
  float* mv   = (float*)alloc((size_t)2 * 1024 * 4);
  _Float16* w1  = (_Float16*)alloc((size_t)110592 * 2);
  _Float16* w2  = (_Float16*)alloc((size_t)32768 * 2);
  _Float16* w3  = (_Float16*)alloc((size_t)221184 * 2);
  _Float16* w4  = (_Float16*)alloc((size_t)262144 * 2);
  _Float16* w5  = (_Float16*)alloc((size_t)3538944 * 2);
  _Float16* wd1 = (_Float16*)alloc((size_t)4194304 * 2);
  _Float16* wd2 = (_Float16*)alloc((size_t)8388608 * 2);
  _Float16* wf1 = (_Float16*)alloc((size_t)8388608 * 2);

  // weights -> f16, transposed to Bt[Co][K]
  k_cvt_f16_t<<<(110592 + 255) / 256, 256, 0, stream>>>(SCW[0], w1, 1728, 64);
  k_cvt_f16_t<<<(32768 + 255) / 256, 256, 0, stream>>>(SCW[1], w2, 512, 64);
  k_cvt_f16_t<<<(221184 + 255) / 256, 256, 0, stream>>>(SCW[2], w3, 1728, 128);
  k_cvt_f16_t<<<(262144 + 255) / 256, 256, 0, stream>>>(SCW[3], w4, 1024, 256);
  k_cvt_f16_t<<<(3538944 + 255) / 256, 256, 0, stream>>>(SCW[4], w5, 6912, 512);
  k_cvt_f16_t<<<(4194304 + 255) / 256, 256, 0, stream>>>(DC1W, wd1, 4096, 1024);
  k_cvt_f16_t<<<(8388608 + 255) / 256, 256, 0, stream>>>(DC2W, wd2, 8192, 1024);
  k_cvt_f16_t<<<(8388608 + 255) / 256, 256, 0, stream>>>(FC1W, wf1, 8192, 1024);

  // point branch
  k_posenc<<<(NP + 255) / 256, 256, 0, stream>>>(X, f0);
  k_knn<<<(NP + 255) / 256, 256, 0, stream>>>(X, kidx, kd2);
  k_pc1<<<NP, 64, 0, stream>>>(f0, kidx, PC1W, PC1B, LN1G, LN1B, f1);
  k_pc2<<<NP, 64, 0, stream>>>(f1, kidx, kd2, PC2W, PC2B, LN2G, LN2B, f2);

  // voxelize (mean)
  hipMemsetAsync(v40, 0, (size_t)NV40 * 64 * 4, stream);
  hipMemsetAsync(cnt, 0, (size_t)NV40 * 4, stream);
  k_voxelize<<<NP, 64, 0, stream>>>(X, f2, v40, cnt);
  k_vox_final<<<(NV40 * 64 + 255) / 256, 256, 0, stream>>>(v40, cnt, m40, NV40);

  // sc1: 40^3, 64->64, k3 s1 SAME
  k_conv_wmma<<<dim3(NV40 / 32, 1), 256, 0, stream>>>(v40, w1, SCB[0], y40, 40, 40, 64, 64, 3, 1, 1);
  k_mask_dilate<<<(NV40 + 255) / 256, 256, 0, stream>>>(m40, m40b, 40, 40, 3, 1, 1);
  k_ln_relu_mask<<<NV40, 128, 0, stream>>>(y40, m40b, SLG[0], SLB[0], 64);
  // sc2: 40->20, 64->64, k2 s2 VALID
  k_conv_wmma<<<dim3(NV20 / 32, 1), 256, 0, stream>>>(y40, w2, SCB[1], y20, 40, 20, 64, 64, 2, 2, 0);
  k_mask_dilate<<<(NV20 + 255) / 256, 256, 0, stream>>>(m40b, m20a, 40, 20, 2, 2, 0);
  k_ln_relu_mask<<<NV20, 128, 0, stream>>>(y20, m20a, SLG[1], SLB[1], 64);
  // sc3: 20^3, 64->128, k3 s1 SAME
  k_conv_wmma<<<dim3(NV20 / 32, 2), 256, 0, stream>>>(y20, w3, SCB[2], y20b, 20, 20, 64, 128, 3, 1, 1);
  k_mask_dilate<<<(NV20 + 255) / 256, 256, 0, stream>>>(m20a, m20b, 20, 20, 3, 1, 1);
  k_ln_relu_mask<<<NV20, 128, 0, stream>>>(y20b, m20b, SLG[2], SLB[2], 128);
  // sc4: 20->10, 128->256, k2 s2 VALID
  k_conv_wmma<<<dim3(NV10 / 32, 4), 256, 0, stream>>>(y20b, w4, SCB[3], y10a, 20, 10, 128, 256, 2, 2, 0);
  k_mask_dilate<<<(NV10 + 255) / 256, 256, 0, stream>>>(m20b, m10a, 20, 10, 2, 2, 0);
  k_ln_relu_mask<<<NV10, 128, 0, stream>>>(y10a, m10a, SLG[3], SLB[3], 256);
  // sc5: 10^3, 256->512, k3 s1 SAME
  k_conv_wmma<<<dim3(NV10 / 32, 8), 256, 0, stream>>>(y10a, w5, SCB[4], y10b, 10, 10, 256, 512, 3, 1, 1);
  k_mask_dilate<<<(NV10 + 255) / 256, 256, 0, stream>>>(m10a, m10b, 10, 10, 3, 1, 1);
  k_ln_relu_mask<<<NV10, 128, 0, stream>>>(y10b, m10b, SLG[4], SLB[4], 512);

  // dense tail
  k_conv_wmma<<<dim3((ND1 + 31) / 32, 16), 256, 0, stream>>>(y10b, wd1, DC1B, yd1, 10, 5, 512, 1024, 2, 2, 0);
  k_bn_stats<<<1024, 256, 0, stream>>>(yd1, mv, ND1, 1024);
  k_bn_apply_relu<<<(ND1 * 1024 + 255) / 256, 256, 0, stream>>>(yd1, mv, BN1G, BN1B, ND1, 1024);
  k_conv_wmma<<<dim3((ND2 + 31) / 32, 16), 256, 0, stream>>>(yd1, wd2, DC2B, yd2, 5, 2, 1024, 1024, 2, 2, 0);
  k_bn_stats<<<1024, 256, 0, stream>>>(yd2, mv, ND2, 1024);
  k_bn_apply_relu<<<(ND2 * 1024 + 255) / 256, 256, 0, stream>>>(yd2, mv, BN2G, BN2B, ND2, 1024);
  // fc1 on flattened [8, 8192] (NDHWC flatten == reference reshape)
  k_gemm_wmma<<<64, 32, 0, stream>>>(yd2, wf1, FC1B, yf1, BATCH, 8192, 1024);
  k_bn_stats<<<1024, 256, 0, stream>>>(yf1, mv, BATCH, 1024);
  k_bn_apply_relu<<<(BATCH * 1024 + 255) / 256, 256, 0, stream>>>(yf1, mv, BN3G, BN3B, BATCH, 1024);
  // fc2 -> output [8, 40]
  k_fc2<<<(BATCH * 40 + 63) / 64, 64, 0, stream>>>(yf1, FC2W, FC2B, (float*)d_out);
}
